// MockRouter_76192719831317
// MI455X (gfx1250) — matrix operand order repaired and checked by reading, not compiled, not run on hardware
//
#include <hip/hip_runtime.h>
#include <hip/hip_bf16.h>

// MoE router: logits = x @ gate_w^T (bf16 -> f32 via WMMA), sigmoid, +bias,
// top-2 (lowest-index tie break), normalized weights (bf16) + indices (i32).
//
// Roofline: x is 1 GiB streamed once -> ~46us at 23.3 TB/s; compute is ~8.6
// GFLOP, negligible. Design: all 16 B128 NT loads of a 16-token tile issued
// before the WMMA chain (sched_barrier-enforced -> 16 outstanding loads per
// wave), then 8x WMMA 16x16x32 bf16; gate weights register-resident.

typedef __bf16 bf16_t;
typedef __attribute__((ext_vector_type(16))) __bf16 v16bf;
typedef __attribute__((ext_vector_type(8)))  __bf16 v8bf;
typedef __attribute__((ext_vector_type(8)))  float  v8f;

#define ROUTER_DIM    256
#define ROUTER_NEXP   8
#define ROUTER_TOPK   2
#define TILE_TOKENS   16
#define KSTEPS        (ROUTER_DIM / 32)      // 8 WMMA k-steps
#define WAVES_PER_BLK 8
#define BLK_THREADS   (WAVES_PER_BLK * 32)   // wave32

__global__ __launch_bounds__(BLK_THREADS)
void moe_router_wmma_kernel(const bf16_t* __restrict__ x,
                            const bf16_t* __restrict__ gw,
                            const float*  __restrict__ bias,
                            bf16_t* __restrict__ w_out,
                            int*    __restrict__ i_out,
                            int ntokens, int ntiles)
{
    __shared__ float lds_logits[WAVES_PER_BLK][TILE_TOKENS * ROUTER_NEXP];

    const int lane = threadIdx.x & 31;
    const int wave = threadIdx.x >> 5;
    const int half = lane >> 4;   // 0: lanes 0-15, 1: lanes 16-31
    const int sub  = lane & 15;
    const int koff = half * 16;   // A-matrix K: contiguous 16 per lane half (matches B)

    // ---- Preload B: gate_w as 32x16 (KxN) tiles for all 8 K-steps. ----
    // Lane = expert column (experts 8..15 are zero padding). In this code's
    // k order (same as A), each lane holds 16 contiguous K values (lanes 0-15: K=s*32+0..15,
    // lanes 16-31: K=s*32+16..31) packed 2-per-VGPR -> one contiguous 32B load.
    v16bf bmat[KSTEPS];
    {
        v16bf zb;
        #pragma unroll
        for (int i = 0; i < 16; ++i) zb[i] = (__bf16)0.0f;
        #pragma unroll
        for (int s = 0; s < KSTEPS; ++s) bmat[s] = zb;
        if (sub < ROUTER_NEXP) {
            const bf16_t* gp = gw + sub * ROUTER_DIM + half * 16;
            #pragma unroll
            for (int s = 0; s < KSTEPS; ++s)
                bmat[s] = *(const v16bf*)(gp + s * 32);
        }
    }
    float bb[ROUTER_NEXP];
    #pragma unroll
    for (int e = 0; e < ROUTER_NEXP; ++e) bb[e] = bias[e];

    const int waveId = blockIdx.x * WAVES_PER_BLK + wave;
    const int nwaves = gridDim.x * WAVES_PER_BLK;

    for (int tile = waveId; tile < ntiles; tile += nwaves) {
        const long t0  = (long)tile * TILE_TOKENS;
        long       tok = t0 + sub;
        if (tok >= ntokens) tok = ntokens - 1;   // tail clamp (redundant work)
        const bf16_t* rowp = x + tok * (long)ROUTER_DIM;

        // ---- stage 1: issue ALL 16 B128 loads for this tile. The
        // sched_barrier below pins them ahead of the WMMA chain so each wave
        // keeps 16 loads in flight (memory-level parallelism). ----
        v8bf a0[KSTEPS], a1[KSTEPS];
        #pragma unroll
        for (int s = 0; s < KSTEPS; ++s) {
            a0[s] = __builtin_nontemporal_load((const v8bf*)(rowp + s * 32 + koff));
            a1[s] = __builtin_nontemporal_load((const v8bf*)(rowp + s * 32 + 8 + koff));
        }
        __builtin_amdgcn_sched_barrier(0);   // no motion across: loads stay hoisted

        // ---- stage 2: 16 tokens x 16 experts logits via 8x WMMA bf16.
        // Loads return in order, so waits before each WMMA are partial. ----
        v8f c = {0.f, 0.f, 0.f, 0.f, 0.f, 0.f, 0.f, 0.f};
        #pragma unroll
        for (int s = 0; s < KSTEPS; ++s) {
            v16bf a = __builtin_shufflevector(a0[s], a1[s],
                0, 1, 2, 3, 4, 5, 6, 7, 8, 9, 10, 11, 12, 13, 14, 15);
            c = __builtin_amdgcn_wmma_f32_16x16x32_bf16(
                    false, a, false, bmat[s], (short)0, c, false, false);
        }

        // ---- scatter logits to LDS: C layout lane&15 = expert, VGPR r ->
        // token half*8 + r. Only expert lanes 0..7 (per half) are valid. ----
        if (sub < ROUTER_NEXP) {
            float* l = &lds_logits[wave][0];
            #pragma unroll
            for (int r = 0; r < 8; ++r)
                l[(half * 8 + r) * ROUTER_NEXP + sub] = c[r];
        }
        // Same-wave LDS ops are in-order; wait for store data + block compiler
        // reordering before the cross-lane gather.
        asm volatile("s_wait_dscnt 0" ::: "memory");
        __builtin_amdgcn_wave_barrier();

        // ---- per-token epilogue: one token per lane (lanes 0-15) ----
        if (lane < TILE_TOKENS) {
            long t = t0 + lane;
            if (t < ntokens) {
                const float* l = &lds_logits[wave][lane * ROUTER_NEXP];
                float bestV = -INFINITY, secV = -INFINITY;
                float bestS = 0.f, secS = 0.f;
                int   bestI = 0, secI = 0;
                #pragma unroll
                for (int e = 0; e < ROUTER_NEXP; ++e) {
                    float sc = 1.0f / (1.0f + __expf(-l[e]));   // sigmoid
                    float v  = sc + bb[e];                      // selection score
                    if (v > bestV) {
                        secV = bestV; secS = bestS; secI = bestI;
                        bestV = v;    bestS = sc;   bestI = e;
                    } else if (v > secV) {
                        secV = v; secS = sc; secI = e;
                    }
                }
                float denom = bestS + secS;
                denom = denom < 1e-12f ? 1e-12f : denom;
                float w0 = bestS / denom;
                float w1 = secS  / denom;

                union { __bf16 h[2]; unsigned u; } pw;
                pw.h[0] = (__bf16)w0;
                pw.h[1] = (__bf16)w1;
                __builtin_nontemporal_store(pw.u, (unsigned*)(w_out + t * ROUTER_TOPK));

                unsigned long long pi =
                    ((unsigned long long)(unsigned)secI << 32) | (unsigned)bestI;
                __builtin_nontemporal_store(pi,
                    (unsigned long long*)(i_out + t * ROUTER_TOPK));
            }
        }
        __builtin_amdgcn_wave_barrier();   // protect LDS before next tile
    }
}

extern "C" void kernel_launch(void* const* d_in, const int* in_sizes, int n_in,
                              void* d_out, int out_size, void* d_ws, size_t ws_size,
                              hipStream_t stream) {
    const bf16_t* x    = (const bf16_t*)d_in[0];   // [T, 256] bf16
    const bf16_t* gw   = (const bf16_t*)d_in[1];   // [8, 256] bf16
    const float*  bias = (const float*)d_in[2];    // [8] f32

    const int ntokens = in_sizes[0] / ROUTER_DIM;
    const int ntiles  = (ntokens + TILE_TOKENS - 1) / TILE_TOKENS;

    // Output layout: weights [T,2] bf16, then indices [T,2] int32,
    // byte-concatenated in reference return order.
    bf16_t* w_out = (bf16_t*)d_out;
    int*    i_out = (int*)((char*)d_out +
                           (size_t)ntokens * ROUTER_TOPK * sizeof(bf16_t));

    int blocks = (ntiles + WAVES_PER_BLK - 1) / WAVES_PER_BLK;
    if (blocks > 4096) blocks = 4096;   // ~4 tiles per wave amortizes B preload
    if (blocks < 1)    blocks = 1;

    moe_router_wmma_kernel<<<blocks, BLK_THREADS, 0, stream>>>(
        x, gw, bias, w_out, i_out, ntokens, ntiles);
}
